// LLaDA2MoeGate_506806141421
// MI455X (gfx1250) — matrix operand order repaired and checked by reading, not compiled, not run on hardware
//
#include <hip/hip_runtime.h>
#include <math.h>

typedef __attribute__((ext_vector_type(16))) __bf16 v16bf;
typedef __attribute__((ext_vector_type(8)))  __bf16 v8bf;
typedef __attribute__((ext_vector_type(8)))  float  v8f;

#define H_DIM 4096
#define E_DIM 256
#define TOKENS_PER_BLOCK 64
#define NTILES 8          // expert tiles per wave (8 x 16 = 128 experts)
#define ROWF 264          // padded LDS row stride (floats) to avoid bank conflicts
#define TOPK 8

// sched_group_barrier masks: 0x2 = VALU, 0x8 = MFMA/WMMA, 0x20 = VMEM read
#if defined(__has_builtin)
# if __has_builtin(__builtin_amdgcn_sched_group_barrier)
#  define SGB(m, s, id) __builtin_amdgcn_sched_group_barrier((m), (s), (id))
# endif
#endif
#ifndef SGB
# define SGB(m, s, id)
#endif

__device__ __forceinline__ float sigmoid_fast(float x) {
  return 1.0f / (1.0f + __expf(-x));
}

// ---------------- Pass 1: fp32 weight -> bf16 weight (runs once per call, 2MB) ----------
__global__ __launch_bounds__(256)
void cvt_weight_kernel(const float* __restrict__ w, unsigned short* __restrict__ wbf_us) {
  __bf16* wbf = (__bf16*)wbf_us;
  const size_t i = ((size_t)blockIdx.x * blockDim.x + threadIdx.x) * 8;
  const float4 f0 = *(const float4*)(w + i);
  const float4 f1 = *(const float4*)(w + i + 4);
  v8bf o;
  o[0] = (__bf16)f0.x; o[1] = (__bf16)f0.y; o[2] = (__bf16)f0.z; o[3] = (__bf16)f0.w;
  o[4] = (__bf16)f1.x; o[5] = (__bf16)f1.y; o[6] = (__bf16)f1.z; o[7] = (__bf16)f1.w;
  *(v8bf*)(wbf + i) = o;
}

// ---------------- Pass 2: router GEMM (bf16 WMMA) + group-limited top-k routing --------
// 256 threads = 8 waves. Wave w: tokens [(w>>1)*16, +16), experts [(w&1)*128, +128).
__global__ __launch_bounds__(256)
void moe_gate_kernel(const float* __restrict__ hid,
                     const unsigned short* __restrict__ wbf_us,
                     const float* __restrict__ bias,
                     float* __restrict__ outW,
                     int* __restrict__ outI) {
  const __bf16* wb = (const __bf16*)wbf_us;

  extern __shared__ float lds[];
  float* scoreRows = lds;                            // 64 * 264 floats
  float* biasLds   = lds + TOKENS_PER_BLOCK * ROWF;  // 256 floats

  const int tid  = threadIdx.x;
  const int lane = tid & 31;
  const int wv   = tid >> 5;        // wave id 0..7
  const int m    = lane & 15;       // row (A) / col (B) within 16x16 tile
  const int hiH  = lane >> 4;       // upper half-wave flag
  const int tRow = wv >> 1;         // token tile row 0..3
  const int eHalf = (wv & 1) * 128; // expert half base

  if (tid < E_DIM) biasLds[tid] = bias[tid];

  // ---------------- GEMM phase: 16 tokens x 128 experts per wave ----------------
  const int tBase = blockIdx.x * TOKENS_PER_BLOCK + tRow * 16;
  // A k order used here (contiguous, same as B): lanes 0-15 (M=lane) hold K 0..15; lanes 16-31 hold K 16..31
  const float* pA = hid + (size_t)(tBase + m) * H_DIM + hiH * 16;
  // B k order used here (contiguous, same as A): lanes 0-15 (N=lane) hold K 0..15; lanes 16-31 hold K 16..31
  const __bf16* pB = wb + (size_t)(eHalf + m) * H_DIM + hiH * 16;

  v8f acc[NTILES];
  #pragma unroll
  for (int n = 0; n < NTILES; ++n) {
    #pragma unroll
    for (int r = 0; r < 8; ++r) acc[n][r] = 0.0f;
  }

  // --- Loop-carried software pipeline state ---
  // B ring buffer (depth 3): one-time prologue loads tiles 0..2 of k=0; thereafter
  // slots n=5,6,7 of each iteration prefetch tiles 0..2 of the NEXT k chunk.
  v16bf bbuf[4];
  #pragma unroll
  for (int p = 0; p < 3; ++p) {
    bbuf[p] = *(const v16bf*)(pB + (size_t)p * (16 * H_DIM));
  }
  // A double buffer across k: 4x float4 of the current chunk, loaded one iteration ahead.
  float4 a0 = *(const float4*)(pA);
  float4 a1 = *(const float4*)(pA + 4);
  float4 a2 = *(const float4*)(pA + 8);
  float4 a3 = *(const float4*)(pA + 12);

  for (int k0 = 0; k0 < H_DIM; k0 += 32) {
    // next k chunk; wraps to 0 on the last iteration (harmless in-bounds reload)
    const int kn = (k0 + 32) & (H_DIM - 1);

    // Convert the A chunk loaded one iteration ago (no load latency on this path).
    v16bf af;
    af[0]  = (__bf16)a0.x; af[1]  = (__bf16)a0.y; af[2]  = (__bf16)a0.z; af[3]  = (__bf16)a0.w;
    af[4]  = (__bf16)a1.x; af[5]  = (__bf16)a1.y; af[6]  = (__bf16)a1.z; af[7]  = (__bf16)a1.w;
    af[8]  = (__bf16)a2.x; af[9]  = (__bf16)a2.y; af[10] = (__bf16)a2.z; af[11] = (__bf16)a2.w;
    af[12] = (__bf16)a3.x; af[13] = (__bf16)a3.y; af[14] = (__bf16)a3.z; af[15] = (__bf16)a3.w;

    // Issue next iteration's A loads (consumed after 8 WMMAs -> latency fully hidden).
    a0 = *(const float4*)(pA + kn);
    a1 = *(const float4*)(pA + kn + 4);
    a2 = *(const float4*)(pA + kn + 8);
    a3 = *(const float4*)(pA + kn + 12);

    #pragma unroll
    for (int n = 0; n < NTILES; ++n) {
      // Prefetch 3 tiles ahead; positions 5..7 fetch tiles 0..2 of the next k chunk,
      // keeping the ring full across the loop boundary.
      if (n + 3 < NTILES) {
        bbuf[(n + 3) & 3] = *(const v16bf*)(pB + (size_t)(n + 3) * (16 * H_DIM) + k0);
      } else {
        bbuf[(n + 3) & 3] = *(const v16bf*)(pB + (size_t)(n + 3 - NTILES) * (16 * H_DIM) + kn);
      }
      acc[n] = __builtin_amdgcn_wmma_f32_16x16x32_bf16(
          /*neg_a=*/false, af, /*neg_b=*/false, bbuf[n & 3],
          /*c_mod=*/(short)0, acc[n], /*reuse_a=*/false, /*reuse_b=*/false);
    }

    // Pipeline shape: [8 cvt VALU] [4 A vmem reads] ([2 B vmem reads][1 WMMA]) x8
    SGB(0x002, 8, 0);
    SGB(0x020, 4, 0);
    SGB(0x020, 2, 0); SGB(0x008, 1, 0);
    SGB(0x020, 2, 0); SGB(0x008, 1, 0);
    SGB(0x020, 2, 0); SGB(0x008, 1, 0);
    SGB(0x020, 2, 0); SGB(0x008, 1, 0);
    SGB(0x020, 2, 0); SGB(0x008, 1, 0);
    SGB(0x020, 2, 0); SGB(0x008, 1, 0);
    SGB(0x020, 2, 0); SGB(0x008, 1, 0);
    SGB(0x020, 2, 0); SGB(0x008, 1, 0);
  }

  // C layout: lanes 0-15 hold N=lane, M=reg (0..7); lanes 16-31 hold M=reg+8
  #pragma unroll
  for (int n = 0; n < NTILES; ++n) {
    const int col = eHalf + n * 16 + m;
    float* dst = scoreRows + (size_t)(tRow * 16 + hiH * 8) * ROWF + col;
    #pragma unroll
    for (int r = 0; r < 8; ++r) dst[r * ROWF] = acc[n][r];
  }
  __syncthreads();

  // ---------------- Routing phase: one thread per token ----------------
  if (tid < TOKENS_PER_BLOCK) {
    float* row = scoreRows + (size_t)tid * ROWF;

    // sigmoid (stored back) + per-group top-2 sum of (sigmoid + bias)
    float gscore[8];
    #pragma unroll
    for (int g = 0; g < 8; ++g) {
      float m1 = -3.0e38f, m2 = -3.0e38f;
      for (int j = 0; j < 32; ++j) {
        const int e = g * 32 + j;
        const float s = sigmoid_fast(row[e]);
        row[e] = s;
        const float r = s + biasLds[e];
        if (r > m1) { m2 = m1; m1 = r; }
        else if (r > m2) { m2 = r; }
      }
      gscore[g] = m1 + m2;
    }

    // top-4 groups (lowest index wins ties, matching jax top_k)
    int gmask = 0;
    for (int rank = 0; rank < 4; ++rank) {
      float best = -3.0e38f; int bi = 0;
      #pragma unroll
      for (int g = 0; g < 8; ++g) {
        const bool sel = (((gmask >> g) & 1) == 0) && (gscore[g] > best);
        best = sel ? gscore[g] : best;
        bi   = sel ? g : bi;
      }
      gmask |= (1 << bi);
    }

    // top-8 experts within selected groups via unrolled sorted insertion
    float vals[TOPK]; int idxs[TOPK];
    #pragma unroll
    for (int i = 0; i < TOPK; ++i) { vals[i] = -3.0e38f; idxs[i] = 0; }
    for (int g = 0; g < 8; ++g) {
      if (((gmask >> g) & 1) == 0) continue;
      for (int j = 0; j < 32; ++j) {
        const int e = g * 32 + j;
        const float r = row[e] + biasLds[e];
        if (r > vals[TOPK - 1]) {
          vals[TOPK - 1] = r; idxs[TOPK - 1] = e;
          #pragma unroll
          for (int q = TOPK - 1; q > 0; --q) {
            if (vals[q] > vals[q - 1]) {
              const float tv = vals[q]; vals[q] = vals[q - 1]; vals[q - 1] = tv;
              const int   ti = idxs[q]; idxs[q] = idxs[q - 1]; idxs[q - 1] = ti;
            }
          }
        }
      }
    }

    // gather sigmoid scores, normalize, scale
    float wsel[TOPK];
    float wsum = 1e-20f;
    #pragma unroll
    for (int i = 0; i < TOPK; ++i) { wsel[i] = row[idxs[i]]; wsum += wsel[i]; }
    const float scale = 2.5f / wsum;

    const size_t gt = (size_t)blockIdx.x * TOKENS_PER_BLOCK + tid;
    #pragma unroll
    for (int i = 0; i < TOPK; ++i) {
      outW[gt * TOPK + i] = wsel[i] * scale;
      outI[gt * TOPK + i] = idxs[i];
    }
  }
}

extern "C" void kernel_launch(void* const* d_in, const int* in_sizes, int n_in,
                              void* d_out, int out_size, void* d_ws, size_t ws_size,
                              hipStream_t stream) {
  const float* hid  = (const float*)d_in[0];   // [T, 4096] f32
  const float* wgt  = (const float*)d_in[1];   // [256, 4096] f32
  const float* bias = (const float*)d_in[2];   // [256] f32
  const int T = in_sizes[0] / H_DIM;           // 16384

  float* outW = (float*)d_out;                             // [T, 8] f32 weights
  int*   outI = (int*)d_out + (size_t)T * TOPK;            // [T, 8] i32 indices (concatenated)

  unsigned short* wbf = (unsigned short*)d_ws;             // [256, 4096] bf16 (2 MB)

  // Pass 1: convert weight to bf16 once
  {
    dim3 grid((E_DIM * H_DIM) / (8 * 256));
    dim3 block(256);
    hipLaunchKernelGGL(cvt_weight_kernel, grid, block, 0, stream, wgt, wbf);
  }

  // Pass 2: fused router GEMM + routing
  {
    const size_t ldsBytes = (size_t)(TOKENS_PER_BLOCK * ROWF + E_DIM) * sizeof(float);
    dim3 grid(T / TOKENS_PER_BLOCK);
    dim3 block(256);
    hipLaunchKernelGGL(moe_gate_kernel, grid, block, ldsBytes, stream,
                       hid, wbf, bias, outW, outI);
  }
}
